// QuantumModelv2_67336497267002
// MI455X (gfx1250) — matrix operand order repaired and checked by reading, not compiled, not run on hardware
//
#include <hip/hip_runtime.h>

#define NL      8
#define DD      64
#define BATCH   512
#define HEAD_IN 8192
#define HID     4096

typedef __attribute__((ext_vector_type(2)))  float  v2f;
typedef __attribute__((ext_vector_type(8)))  float  v8f;
typedef __attribute__((ext_vector_type(8)))  __bf16 v8bf;
typedef __attribute__((ext_vector_type(16))) __bf16 v16bf;

// ---------------------------------------------------------------------------
// Complex 64x64 chain: one block per batch element, 256 threads = 8 waves.
// f32 WMMA 16x16x4. LDS ping-pong between (cr,ci) and (tr,ti).
// ---------------------------------------------------------------------------
__device__ __forceinline__ void cmm64(const float* __restrict__ Ar,
                                      const float* __restrict__ Ai,
                                      const float* __restrict__ Sr,
                                      const float* __restrict__ Si,
                                      float* __restrict__ Dr,
                                      float* __restrict__ Di)
{
  const int tid  = threadIdx.x;
  const int wave = tid >> 5;
  const int lane = tid & 31;
  const int m    = lane & 15;
  const int kofs = (lane >> 4) << 1;   // 0 or 2 (K sub-offset per half-wave)
  const int t0   = wave * 2;           // two adjacent tiles, same tile row
  const int tm   = t0 >> 2;
  const int tn0  = t0 & 3;
  const int tn1  = tn0 + 1;

  v8f p0 = {}, q0 = {}, i0 = {};
  v8f p1 = {}, q1 = {}, i1 = {};

  const int arow = tm * 16 + m;
  const int c0   = tn0 * 16 + m;
  const int c1   = tn1 * 16 + m;

  for (int k = 0; k < 64; k += 4) {
    const int ka = k + kofs;
    v2f ar, ai, br0, bi0, br1, bi1;
    ar.x  = Ar[arow * 64 + ka];      ar.y  = Ar[arow * 64 + ka + 1];
    ai.x  = Ai[arow * 64 + ka];      ai.y  = Ai[arow * 64 + ka + 1];
    br0.x = Sr[ka * 64 + c0];        br0.y = Sr[(ka + 1) * 64 + c0];
    bi0.x = Si[ka * 64 + c0];        bi0.y = Si[(ka + 1) * 64 + c0];
    br1.x = Sr[ka * 64 + c1];        br1.y = Sr[(ka + 1) * 64 + c1];
    bi1.x = Si[ka * 64 + c1];        bi1.y = Si[(ka + 1) * 64 + c1];

    p0 = __builtin_amdgcn_wmma_f32_16x16x4_f32(false, ar, false, br0, (short)0, p0, false, false);
    q0 = __builtin_amdgcn_wmma_f32_16x16x4_f32(false, ai, false, bi0, (short)0, q0, false, false);
    i0 = __builtin_amdgcn_wmma_f32_16x16x4_f32(false, ar, false, bi0, (short)0, i0, false, false);
    i0 = __builtin_amdgcn_wmma_f32_16x16x4_f32(false, ai, false, br0, (short)0, i0, false, false);

    p1 = __builtin_amdgcn_wmma_f32_16x16x4_f32(false, ar, false, br1, (short)0, p1, false, false);
    q1 = __builtin_amdgcn_wmma_f32_16x16x4_f32(false, ai, false, bi1, (short)0, q1, false, false);
    i1 = __builtin_amdgcn_wmma_f32_16x16x4_f32(false, ar, false, bi1, (short)0, i1, false, false);
    i1 = __builtin_amdgcn_wmma_f32_16x16x4_f32(false, ai, false, br1, (short)0, i1, false, false);
  }

  // C/D layout: VGPR v -> M = v + 8*(lane>=16), N = lane&15 (within 16x16 tile)
  const int n0    = tn0 * 16 + (lane & 15);
  const int n1    = tn1 * 16 + (lane & 15);
  const int rbase = tm * 16 + ((lane >> 4) << 3);
#pragma unroll
  for (int v = 0; v < 8; ++v) {
    const int r = rbase + v;
    Dr[r * 64 + n0] = p0[v] - q0[v];
    Di[r * 64 + n0] = i0[v];
    Dr[r * 64 + n1] = p1[v] - q1[v];
    Di[r * 64 + n1] = i1[v];
  }
  __syncthreads();
}

__global__ __launch_bounds__(256)
void chain_kernel(const float* __restrict__ x_r, const float* __restrict__ x_i,
                  const float* __restrict__ u_r, const float* __restrict__ u_i,
                  const float* __restrict__ W_r, const float* __restrict__ W_i,
                  __bf16* __restrict__ feat)
{
  __shared__ float sm[4 * 4096];      // exactly 64 KB
  float* cr = sm;
  float* ci = sm + 4096;
  float* tr = sm + 8192;
  float* ti = sm + 12288;

  const int    b  = blockIdx.x;
  const size_t xb = (size_t)b * NL * 4096;

  for (int t = threadIdx.x; t < 4096; t += 256) {
    cr[t] = x_r[xb + t];
    ci[t] = x_i[xb + t];
  }
  __syncthreads();

  for (int i = 0; i < NL - 1; ++i) {
    cmm64(W_r + (size_t)i * 4096, W_i + (size_t)i * 4096, cr, ci, tr, ti);
    cmm64(x_r + xb + (size_t)(i + 1) * 4096, x_i + xb + (size_t)(i + 1) * 4096,
          tr, ti, cr, ci);
  }
  cmm64(W_r + (size_t)(NL - 1) * 4096, W_i + (size_t)(NL - 1) * 4096, cr, ci, tr, ti);

  // feat[b, p*128 + q] = real, feat[b, p*128 + 64 + q] = imag  of (u - out)
  const size_t ub = (size_t)b * 4096;
  for (int t = threadIdx.x; t < 4096; t += 256) {
    const int p = t >> 6, q = t & 63;
    const float vr = u_r[ub + t] - tr[t];
    const float vi = u_i[ub + t] - ti[t];
    feat[(size_t)b * HEAD_IN + p * 128 + q]      = (__bf16)vr;
    feat[(size_t)b * HEAD_IN + p * 128 + 64 + q] = (__bf16)vi;
  }
}

// ---------------------------------------------------------------------------
// C(MxN) = mish(A(MxK, bf16, row-major) * W(NxK, f32, row-major)^T + bias)
// 64(M) x 128(N) tile per 256-thread block; bf16 WMMA 16x16x32.
// A tile is DMA'd into LDS with GLOBAL_LOAD_ASYNC_TO_LDS_B128 (ASYNCcnt),
// W tile is staged+converted f32->bf16 by the VALU concurrently.
// ---------------------------------------------------------------------------
__device__ __forceinline__ float mish_f(float x) {
  const float sp = (x > 20.0f) ? x : log1pf(__expf(x));
  return x * tanhf(sp);
}

__device__ __forceinline__ v16bf ld_frag_split16(const __bf16* p) {
  // elements [0..7] from p, [8..15] from p+16 (A-frag K layout)
  const v8bf lo = *(const v8bf*)(p);
  const v8bf hi = *(const v8bf*)(p + 16);
  v16bf r;
#pragma unroll
  for (int i = 0; i < 8; ++i) { r[i] = lo[i]; r[i + 8] = hi[i]; }
  return r;
}

__device__ __forceinline__ v16bf ld_frag_contig16(const __bf16* p) {
  // 16 consecutive elements (contiguous K order; unused)
  const v8bf lo = *(const v8bf*)(p);
  const v8bf hi = *(const v8bf*)(p + 8);
  v16bf r;
#pragma unroll
  for (int i = 0; i < 8; ++i) { r[i] = lo[i]; r[i + 8] = hi[i]; }
  return r;
}

__global__ __launch_bounds__(256)
void gemm_mish_kernel(const __bf16* __restrict__ A,
                      const float*  __restrict__ Wf,
                      const float*  __restrict__ bias,
                      __bf16* __restrict__ C,
                      int M, int N, int K)
{
  __shared__ __bf16 Bs[128 * 40];     // weight tile, padded pitch 40
  __shared__ __bf16 As[64 * 40];      // activation tile, padded pitch 40

  const int tid  = threadIdx.x;
  const int wave = tid >> 5;
  const int lane = tid & 31;
  const int bn   = blockIdx.x * 128;
  const int bm   = blockIdx.y * 64;

  const int wm = wave & 1;            // 2 wave rows (M)
  const int wn = wave >> 1;           // 4 wave cols (N)

  const int c0a   = (lane >> 4) << 3;   // A frag sub-K: 0 / 8 (+16 for hi half)
  const int c0b   = (lane >> 4) << 3;   // B frag sub-K: 0 / 8 (+16 for hi half)
  const int arow0 = wm * 32 + (lane & 15);        // row within 64-row tile
  const int arow1 = arow0 + 16;
  const int bcol0 = wn * 32 + (lane & 15);
  const int bcol1 = bcol0 + 16;

  // Async A staging: thread -> one 16B chunk. 64 rows x 4 chunks = 256 threads.
  const int srow = tid >> 2;                      // 0..63
  const int sq   = (tid & 3) * 8;                 // element offset 0/8/16/24
  const unsigned lds_a = (unsigned)(uintptr_t)(&As[srow * 40 + sq]);
  const unsigned gbase = (unsigned)(((size_t)(bm + srow) * K + sq) * 2); // bytes

  v8f acc00 = {}, acc01 = {}, acc10 = {}, acc11 = {};

  for (int kc = 0; kc < K; kc += 32) {
    // 1) kick off async DMA of the 64x32 bf16 A chunk into LDS
    const unsigned goff = gbase + (unsigned)(kc * 2);
    asm volatile("global_load_async_to_lds_b128 %0, %1, %2"
                 :: "v"(lds_a), "v"(goff), "s"(A) : "memory");

    // 2) stage 128x32 chunk of W, converting f32 -> bf16 (coalesced reads)
#pragma unroll
    for (int j = 0; j < 16; ++j) {
      const int idx = tid + j * 256;
      const int r = idx >> 5, c = idx & 31;
      Bs[r * 40 + c] = (__bf16)Wf[(size_t)(bn + r) * K + kc + c];
    }
    if (kc + 32 < K)
      __builtin_prefetch(Wf + (size_t)(bn + (tid >> 1)) * K + kc + 32, 0, 1);

    // 3) wait for this wave's async transfers, then block-wide barrier
    asm volatile("s_wait_asynccnt 0x0" ::: "memory");
    __syncthreads();

    const v16bf a0 = ld_frag_split16(&As[arow0 * 40 + c0a]);
    const v16bf a1 = ld_frag_split16(&As[arow1 * 40 + c0a]);
    const v16bf b0 = ld_frag_split16(&Bs[bcol0 * 40 + c0b]);
    const v16bf b1 = ld_frag_split16(&Bs[bcol1 * 40 + c0b]);

    acc00 = __builtin_amdgcn_wmma_f32_16x16x32_bf16(false, a0, false, b0, (short)0, acc00, false, false);
    acc01 = __builtin_amdgcn_wmma_f32_16x16x32_bf16(false, a0, false, b1, (short)0, acc01, false, false);
    acc10 = __builtin_amdgcn_wmma_f32_16x16x32_bf16(false, a1, false, b0, (short)0, acc10, false, false);
    acc11 = __builtin_amdgcn_wmma_f32_16x16x32_bf16(false, a1, false, b1, (short)0, acc11, false, false);
    __syncthreads();
  }

  const int rb0 = bm + wm * 32 + ((lane >> 4) << 3);
  const int nn0 = bn + wn * 32 + (lane & 15);
  const float bv0 = bias[nn0];
  const float bv1 = bias[nn0 + 16];
#pragma unroll
  for (int v = 0; v < 8; ++v) {
    const int r0 = rb0 + v;
    const int r1 = r0 + 16;
    C[(size_t)r0 * N + nn0]      = (__bf16)mish_f(acc00[v] + bv0);
    C[(size_t)r0 * N + nn0 + 16] = (__bf16)mish_f(acc01[v] + bv1);
    C[(size_t)r1 * N + nn0]      = (__bf16)mish_f(acc10[v] + bv0);
    C[(size_t)r1 * N + nn0 + 16] = (__bf16)mish_f(acc11[v] + bv1);
  }
}

// ---------------------------------------------------------------------------
// out[b] = h2[b,:] . w3 + b3
// ---------------------------------------------------------------------------
__global__ __launch_bounds__(256)
void head_kernel(const __bf16* __restrict__ h2, const float* __restrict__ w3,
                 const float* __restrict__ b3, float* __restrict__ out)
{
  __shared__ float red[256];
  const int b = blockIdx.x;
  float p = 0.0f;
  for (int o = threadIdx.x; o < HID; o += 256)
    p += (float)h2[(size_t)b * HID + o] * w3[o];
  red[threadIdx.x] = p;
  __syncthreads();
  for (int s = 128; s > 0; s >>= 1) {
    if (threadIdx.x < s) red[threadIdx.x] += red[threadIdx.x + s];
    __syncthreads();
  }
  if (threadIdx.x == 0) out[b] = red[0] + b3[0];
}

// ---------------------------------------------------------------------------
extern "C" void kernel_launch(void* const* d_in, const int* in_sizes, int n_in,
                              void* d_out, int out_size, void* d_ws, size_t ws_size,
                              hipStream_t stream)
{
  const float* x_r = (const float*)d_in[0];
  const float* x_i = (const float*)d_in[1];
  const float* u_r = (const float*)d_in[2];
  const float* u_i = (const float*)d_in[3];
  const float* W_r = (const float*)d_in[4];
  const float* W_i = (const float*)d_in[5];
  const float* w1  = (const float*)d_in[6];
  const float* b1  = (const float*)d_in[7];
  const float* w2  = (const float*)d_in[8];
  const float* b2  = (const float*)d_in[9];
  const float* w3  = (const float*)d_in[10];
  const float* b3  = (const float*)d_in[11];
  float* out = (float*)d_out;

  char*   ws   = (char*)d_ws;
  __bf16* feat = (__bf16*)ws;                                        // 8 MB
  __bf16* h1   = (__bf16*)(ws + (size_t)BATCH * HEAD_IN * 2);        // 4 MB
  __bf16* h2   = (__bf16*)(ws + (size_t)BATCH * HEAD_IN * 2
                              + (size_t)BATCH * HID * 2);            // 4 MB

  chain_kernel<<<BATCH, 256, 0, stream>>>(x_r, x_i, u_r, u_i, W_r, W_i, feat);

  gemm_mish_kernel<<<dim3(HID / 128, BATCH / 64), 256, 0, stream>>>(
      feat, w1, b1, h1, BATCH, HID, HEAD_IN);

  gemm_mish_kernel<<<dim3(HID / 128, BATCH / 64), 256, 0, stream>>>(
      h1, w2, b2, h2, BATCH, HID, HID);

  head_kernel<<<BATCH, 256, 0, stream>>>(h2, w3, b3, out);
}
